// DepthRefinement_231928234439
// MI455X (gfx1250) — matrix operand order repaired and checked by reading, not compiled, not run on hardware
//
#include <hip/hip_runtime.h>
#include <cstdint>

typedef _Float16 f16;
typedef __attribute__((ext_vector_type(8)))  _Float16 v8h;
typedef __attribute__((ext_vector_type(16))) _Float16 v16h;
typedef __attribute__((ext_vector_type(8)))  float    v8f;

#define EPS_BN 1e-5f

__device__ __forceinline__ float sigmoidf_(float x){ return 1.f/(1.f+__expf(-x)); }

// ---------------- BN folding: scale = g*rsqrt(v+eps), shift = b + (bias - m)*scale ----
__global__ void k_bnprep(const float* g,const float* b,const float* m,const float* v,
                         const float* bias,float* scale,float* shift,int C,int has_bn){
  int i = blockIdx.x*blockDim.x + threadIdx.x;
  if(i>=C) return;
  if(has_bn){
    float s = g[i]*rsqrtf(v[i]+EPS_BN);
    scale[i] = s;
    shift[i] = b[i] + ((bias? bias[i]:0.f) - m[i])*s;
  }else{
    scale[i] = 1.f;
    shift[i] = bias? bias[i] : 0.f;
  }
}

// ---------------- weight pack: OIHW f32 -> [CoutP][taps][CinP] f16 (zero padded) -----
__global__ void k_pack_w(const float* __restrict__ w,f16* __restrict__ out,
                         int Cout,int Cin,int taps,int CoutP,int CinP){
  long long idx = (long long)blockIdx.x*blockDim.x + threadIdx.x;
  long long total = (long long)CoutP*taps*CinP;
  if(idx>=total) return;
  int c  = (int)(idx % CinP);
  int t  = (int)((idx/CinP) % taps);
  int co = (int)(idx/((long long)CinP*taps));
  float val = (co<Cout && c<Cin) ? w[((long long)co*Cin + c)*taps + t] : 0.f;
  out[idx] = (f16)val;
}

// ---------------- feature pack: NCHW f32 -> [B*HW][CinP] f16 (zero padded channels) --
__global__ void k_pack_feat(const float* __restrict__ src,f16* __restrict__ out,
                            int Bn,int HW,int Csrc,int CinP){
  long long idx = (long long)blockIdx.x*blockDim.x + threadIdx.x;
  long long total = (long long)Bn*HW*CinP;
  if(idx>=total) return;
  int c = (int)(idx % CinP);
  long long pix = idx / CinP;
  int b  = (int)(pix / HW);
  int hw = (int)(pix % HW);
  float v = (c<Csrc) ? src[((long long)b*Csrc + c)*HW + hw] : 0.f;
  out[idx] = (f16)v;
}

// ---------------- implicit-GEMM conv (1x1 or 3x3 pad1) via v_wmma_f32_16x16x32_f16 ---
// feat: [B][H][W][CinP] f16 ; w: [CoutP][taps][CinP] f16 ; out: NCHW f32 at (c0,Ctot)
// Requires Ww % 32 == 0, CinP % 32 == 0. Block = 64 threads (2 waves).
// A tile is DMA'd into LDS with global_load_async_to_lds_b128 (ASYNCcnt), fragments
// read back as two ds_load_b128; B fragments are two b128 global loads per lane.
__global__ __launch_bounds__(64) void k_conv_wmma(
    const f16* __restrict__ feat,const f16* __restrict__ w,
    const float* __restrict__ scale,const float* __restrict__ shift,
    float* __restrict__ out,
    int Bn,int Hh,int Ww,int CinP,int taps,
    int Cout,int Ctot,int c0,int act)
{
  __shared__ __align__(16) f16 aT[16*32];
  const int lane = threadIdx.x & 31;
  const int wv   = threadIdx.x >> 5;
  const int x0   = (blockIdx.x*2 + wv)*16;
  const int y    = blockIdx.y;
  const int CoutT= (Cout+15)>>4;
  const int b    = blockIdx.z / CoutT;
  const int ct   = blockIdx.z % CoutT;
  const int hf   = lane >> 4;
  const int lm   = lane & 15;

  v8f acc = {0.f,0.f,0.f,0.f,0.f,0.f,0.f,0.f};
  const v8h zh = {(f16)0.f,(f16)0.f,(f16)0.f,(f16)0.f,(f16)0.f,(f16)0.f,(f16)0.f,(f16)0.f};
  const long long featB = (long long)b*Hh*Ww*CinP;
  const f16* wct = w + (long long)(ct*16)*taps*CinP;

  // cooperative A-tile fill indices (constant across loop)
  const int e0 = threadIdx.x*8;                 // 8 f16 = 16B per thread
  const int amm = e0>>5, akb = e0&31;
  const unsigned ldsDst = (unsigned)(uintptr_t)(aT + e0);  // LDS byte offset (low 32b)
  // per-lane A fragment LDS addresses (16B aligned)
  const f16* aL0 = aT + lm*32 + 8*hf;
  const f16* aL1 = aT + lm*32 + 16 + 8*hf;

  for(int tap=0; tap<taps; ++tap){
    const int dy = (taps==9)? (tap/3 - 1) : 0;
    const int dx = (taps==9)? (tap%3 - 1) : 0;
    const int yy = y + dy;
    const bool rowok = (yy>=0 && yy<Hh);
    const f16* wrow = wct + (long long)tap*CinP;
    const int xx = x0 + lm + dx;
    const bool ok = rowok && (xx>=0) && (xx<Ww);
    const f16* fp0 = feat + featB + ((long long)yy*Ww + xx)*CinP;

    for(int ck=0; ck<CinP; ck+=32){
      __syncthreads();                          // previous tile consumed by both waves
      {   // A tile: 16 couts x 32 k via async DMA, one b128 per thread
        const f16* gp = wrow + (long long)amm*taps*CinP + ck + akb;
        asm volatile("global_load_async_to_lds_b128 %0, %1, off"
                     :: "v"(ldsDst), "v"(gp) : "memory");
        asm volatile("s_wait_asynccnt 0x0" ::: "memory");
      }
      __syncthreads();                          // tile visible to both waves

      // A fragment: two b128 LDS reads, already in VGPR order
      v8h a0 = *(const v8h*)aL0;
      v8h a1 = *(const v8h*)aL1;
      v16h af = __builtin_shufflevector(a0,a1, 0,1,2,3,4,5,6,7,8,9,10,11,12,13,14,15);

      // B fragment: lane covers k = 8*hf..8*hf+7 and 16+8*hf..16+8*hf+7 -> two b128 loads
      const f16* fp = fp0 + ck;
      v8h b0, b1;
      if(ok){
        b0 = *(const v8h*)(fp + 8*hf); b1 = *(const v8h*)(fp + 16 + 8*hf);
        if((ck+32)<CinP) __builtin_prefetch(fp + 32, 0, 1);
      }else{ b0 = zh; b1 = zh; }
      v16h bf = __builtin_shufflevector(b0,b1, 0,1,2,3,4,5,6,7,8,9,10,11,12,13,14,15);

      acc = __builtin_amdgcn_wmma_f32_16x16x32_f16(
              false, af, false, bf, (short)0, acc, false, false);
    }
  }

  // epilogue: fused BN/bias + activation, C/D layout: m = i + 8*half, n = lane&15
  #pragma unroll
  for(int i=0;i<8;++i){
    int mm = i + 8*hf;
    int co = ct*16 + mm;
    if(co < Cout){
      float v = acc[i]*scale[co] + shift[co];
      if(act==1)      v = fmaxf(v, 0.f);
      else if(act==2) v = sigmoidf_(v);
      out[(((long long)b*Ctot + (c0+co))*Hh + y)*Ww + x0 + lm] = v;
    }
  }
}

// ---------------- SE: global average pool, tiny MLP, channel scale -------------------
__global__ void k_gap(const float* __restrict__ x,float* __restrict__ s,int HW){
  __shared__ float red[256];
  int c = blockIdx.x, b = blockIdx.y, C = gridDim.x;
  const float* p = x + ((long long)b*C + c)*HW;
  float a = 0.f;
  for(int i=threadIdx.x;i<HW;i+=256) a += p[i];
  red[threadIdx.x]=a; __syncthreads();
  for(int st=128; st>0; st>>=1){
    if(threadIdx.x<st) red[threadIdx.x]+=red[threadIdx.x+st];
    __syncthreads();
  }
  if(threadIdx.x==0) s[b*C+c] = red[0]/(float)HW;
}

__global__ void k_se(const float* __restrict__ s,const float* __restrict__ w1,
                     const float* __restrict__ w2,float* __restrict__ o,int comb){
  __shared__ float sv[32], hv[32];
  int b=blockIdx.x, t=threadIdx.x;
  if(t<comb) sv[t]=s[b*comb+t];
  __syncthreads();
  if(t<comb){ float a=0.f; for(int j=0;j<comb;++j) a+=w1[t*comb+j]*sv[j]; hv[t]=fmaxf(a,0.f); }
  __syncthreads();
  if(t<comb){ float a=0.f; for(int j=0;j<comb;++j) a+=w2[t*comb+j]*hv[j]; o[b*comb+t]=sigmoidf_(a); }
}

__global__ void k_chan_scale(float* __restrict__ x,const float* __restrict__ s,
                             int HW,long long total){
  long long i=(long long)blockIdx.x*blockDim.x+threadIdx.x;
  if(i>=total) return;
  x[i] *= s[i/HW];
}

// ---------------- misc elementwise ---------------------------------------------------
__global__ void k_add(float* __restrict__ a,const float* __restrict__ b,long long n){
  long long i=(long long)blockIdx.x*blockDim.x+threadIdx.x;
  if(i<n) a[i]+=b[i];
}
__global__ void k_valid(const float* __restrict__ d,float* __restrict__ v,long long n){
  long long i=(long long)blockIdx.x*blockDim.x+threadIdx.x;
  if(i<n) v[i] = d[i]>0.f ? 1.f : 0.f;
}
__global__ void k_sigmoid_mul(const float* __restrict__ x,const float* __restrict__ v,
                              float* __restrict__ o,long long n){
  long long i=(long long)blockIdx.x*blockDim.x+threadIdx.x;
  if(i<n) o[i] = sigmoidf_(x[i])*v[i];
}
__global__ void k_softmax3(float* __restrict__ x,int HW,int Bn){
  long long i=(long long)blockIdx.x*blockDim.x+threadIdx.x;
  long long total=(long long)Bn*HW;
  if(i>=total) return;
  int b=(int)(i/HW), hw=(int)(i%HW);
  float* p = x + ((long long)b*3)*HW + hw;
  float a=p[0], c=p[HW], d=p[2*HW];
  float mx=fmaxf(a,fmaxf(c,d));
  float e0=__expf(a-mx), e1=__expf(c-mx), e2=__expf(d-mx);
  float inv=1.f/(e0+e1+e2);
  p[0]=e0*inv; p[HW]=e1*inv; p[2*HW]=e2*inv;
}

// ---------------- CSPN affinity normalization ([kk-1] -> [kk] with middle tap) -------
__global__ void k_cspn_norm(const float* __restrict__ g,float* __restrict__ o,
                            int kk,int HW,int Bn){
  long long i=(long long)blockIdx.x*blockDim.x+threadIdx.x;
  long long total=(long long)Bn*HW;
  if(i>=total) return;
  int b=(int)(i/HW), hw=(int)(i%HW);
  const float* gp = g + (long long)b*(kk-1)*HW + hw;
  float s=0.f;
  for(int t=0;t<kk-1;++t) s += fabsf(gp[(long long)t*HW]);
  float inv = 1.f/s;
  int halfk=(kk-1)/2;
  float* op = o + (long long)b*kk*HW + hw;
  float ssum=0.f;
  for(int t=0;t<kk-1;++t){
    float v = gp[(long long)t*HW]*inv; ssum+=v;
    int oc = (t<halfk)? t : t+1;
    op[(long long)oc*HW]=v;
  }
  op[(long long)halfk*HW] = 1.f-ssum;
}

// ---------------- kernel_trans: encoder is one-hot -> pure flip-gather ---------------
__global__ void k_ktrans(const float* __restrict__ in,float* __restrict__ o,
                         int ks,int Hh,int Ww,int Bn){
  int kk=ks*ks, p=(ks-1)/2;
  long long total=(long long)Bn*kk*Hh*Ww;
  long long i=(long long)blockIdx.x*blockDim.x+threadIdx.x;
  if(i>=total) return;
  int x=(int)(i%Ww); long long r=i/Ww;
  int y=(int)(r%Hh); r/=Hh;
  int oc=(int)(r%kk); int b=(int)(r/kk);
  int j=kk-1-oc;
  int yy=y + (ks-1 - j/ks) - p;
  int xx=x + (ks-1 - j%ks) - p;
  float v=0.f;
  if(yy>=0&&yy<Hh&&xx>=0&&xx<Ww) v = in[(((long long)b*kk + j)*Hh + yy)*Ww + xx];
  o[i]=v;
}

// ---------------- SparseDownSampleClose (stride 2) -----------------------------------
__global__ void k_sparse_down(const float* __restrict__ d,const float* __restrict__ valid,
                              float* __restrict__ d2,float* __restrict__ vm,
                              int Hh,int Ww,int Bn){
  int H2=Hh/2, W2=Ww/2;
  long long total=(long long)Bn*H2*W2;
  long long i=(long long)blockIdx.x*blockDim.x+threadIdx.x;
  if(i>=total) return;
  int x=(int)(i%W2); long long r=i/W2;
  int y=(int)(r%H2); int b=(int)(r/H2);
  float emax=-1e30f, vmax=0.f;
  for(int iy=0;iy<2;++iy) for(int ix=0;ix<2;++ix){
    long long idx = ((long long)b*Hh + (2*y+iy))*Ww + (2*x+ix);
    float dv=d[idx], vv=valid[idx];
    emax = fmaxf(emax, -(1.f-vv)*600.f - dv);
    vmax = fmaxf(vmax, vv);
  }
  d2[i] = -emax - (1.f-vmax)*600.f;
  vm[i] = vmax;
}

// ---------------- fused CSPN step + mask blend (optionally half-res g/mask/dref) -----
__global__ void k_cspn_blend(const float* __restrict__ g,const float* __restrict__ src,
                             const float* __restrict__ inp0,const float* __restrict__ mask,
                             const float* __restrict__ dref,float* __restrict__ o,
                             int ks,int dil,int pad,int Hh,int Ww,int Bn,int ghalf){
  long long total=(long long)Bn*Hh*Ww;
  long long i=(long long)blockIdx.x*blockDim.x+threadIdx.x;
  if(i>=total) return;
  int x=(int)(i%Ww); long long r=i/Ww;
  int y=(int)(r%Hh); int b=(int)(r/Hh);
  int kk=ks*ks, mid=(kk-1)/2;
  int H2=Hh/2, W2=Ww/2;
  float acc=0.f;
  for(int t=0;t<kk;++t){
    float kv = ghalf ? g[(((long long)b*kk + t)*H2 + (y>>1))*W2 + (x>>1)]
                     : g[(((long long)b*kk + t)*Hh + y)*Ww + x];
    float sv;
    if(t==mid) sv = inp0[i];
    else{
      int yy=y + (t/ks)*dil - pad, xx=x + (t%ks)*dil - pad;
      sv = (yy>=0&&yy<Hh&&xx>=0&&xx<Ww) ? src[((long long)b*Hh + yy)*Ww + xx] : 0.f;
    }
    acc += kv*sv;
  }
  float mk, dr;
  if(ghalf){
    long long hi=((long long)b*H2 + (y>>1))*W2 + (x>>1);
    mk=mask[hi]; dr=dref[hi];
  }else{ mk=mask[i]; dr=dref[i]; }
  o[i] = mk*dr + (1.f-mk)*acc;
}

// ---------------- 3-way confidence blend (optionally half-res kconf) -----------------
__global__ void k_blend3(const float* __restrict__ kc,const float* __restrict__ d3,
                         const float* __restrict__ d5,const float* __restrict__ d7,
                         float* __restrict__ o,int Hh,int Ww,int Bn,int kchalf){
  long long total=(long long)Bn*Hh*Ww;
  long long i=(long long)blockIdx.x*blockDim.x+threadIdx.x;
  if(i>=total) return;
  int x=(int)(i%Ww); long long r=i/Ww;
  int y=(int)(r%Hh); int b=(int)(r/Hh);
  float w0,w1,w2;
  if(kchalf){
    int H2=Hh/2, W2=Ww/2; long long hw=(long long)H2*W2;
    long long bb=((long long)b*3)*hw + (long long)(y>>1)*W2 + (x>>1);
    w0=kc[bb]; w1=kc[bb+hw]; w2=kc[bb+2*hw];
  }else{
    long long hw=(long long)Hh*Ww;
    long long bb=((long long)b*3)*hw + (long long)y*Ww + x;
    w0=kc[bb]; w1=kc[bb+hw]; w2=kc[bb+2*hw];
  }
  o[i] = w0*d3[i] + w1*d5[i] + w2*d7[i];
}

// =====================================================================================
extern "C" void kernel_launch(void* const* d_in,const int* in_sizes,int n_in,
                              void* d_out,int out_size,void* d_ws,size_t ws_size,
                              hipStream_t stream){
  (void)in_sizes; (void)n_in; (void)out_size; (void)ws_size;
  const int BB=2, HH=352, WW=1216, H2=176, W2=608;
  const long long HW1=(long long)HH*WW, HW2=(long long)H2*W2;
  const long long P1=BB*HW1, P2=BB*HW2;

  auto F = [&](int i){ return (const float*)d_in[i]; };
  auto blk = [](long long n){ return (int)((n+255)/256); };

  // ---- workspace bump allocation ----
  char* base=(char*)d_ws; size_t off=0;
  auto alloc=[&](size_t bytes)->char*{ size_t o=(off+255)&~(size_t)255; off=o+bytes; return base+o; };
  float* feat1 =(float*)alloc((size_t)BB*64 *HW1*4);
  float* feat2 =(float*)alloc((size_t)BB*128*HW2*4);
  f16*   feat1h=(f16*)  alloc((size_t)BB*HW1*64 *2);
  f16*   feat2h=(f16*)  alloc((size_t)BB*HW2*128*2);
  float* g3b=(float*)alloc((size_t)BB*9 *HW1*4);
  float* g5b=(float*)alloc((size_t)BB*25*HW1*4);
  float* g7b=(float*)alloc((size_t)BB*49*HW1*4);
  float* g3h=(float*)alloc((size_t)BB*9 *HW2*4);
  float* g5h=(float*)alloc((size_t)BB*25*HW2*4);
  float* g7h=(float*)alloc((size_t)BB*49*HW2*4);
  float* validb=(float*)alloc((size_t)P1*4);
  float* maskb =(float*)alloc((size_t)P1*4);
  float* kconf =(float*)alloc((size_t)BB*3*HW1*4);
  float* mask2b=(float*)alloc((size_t)P2*4);
  float* kconf2=(float*)alloc((size_t)BB*3*HW2*4);
  float* ds2=(float*)alloc((size_t)P2*4);
  float* vm2=(float*)alloc((size_t)P2*4);
  float* dep2=(float*)alloc((size_t)P1*4);
  float* t3=(float*)alloc((size_t)P1*4);
  float* t5=(float*)alloc((size_t)P1*4);
  float* t7=(float*)alloc((size_t)P1*4);
  f16*   wslab=(f16*)alloc(1<<17);
  float* scsl=(float*)alloc(64*4);
  float* shsl=(float*)alloc(64*4);
  float* sev=(float*)alloc((size_t)BB*32*4);
  float* seo=(float*)alloc((size_t)BB*32*4);
  char*  SCR = alloc(340ull*1000*1000);     // time-shared scratch region
  float* sA  =(float*)(SCR);                //  GUB phase
  float* sXY =(float*)(SCR+ 60ll*1000*1000);
  float* sB  =(float*)(SCR+120ll*1000*1000);
  f16*   fsl =(f16*)  (SCR+180ll*1000*1000);
  float* rawS=(float*)(SCR);                //  CSPN phase (after GUBs complete)
  float* normS=(float*)(SCR+170ll*1000*1000);

  auto packf=[&](const float* src,f16* dst,int Bn,long long HW,int Csrc,int CinP){
    long long total=(long long)Bn*HW*CinP;
    k_pack_feat<<<blk(total),256,0,stream>>>(src,dst,Bn,(int)HW,Csrc,CinP);
  };
  auto conv=[&](const f16* feat,int CinP,const float* wsrc,const float* bias,
                const float* bng,const float* bnb,const float* bnm,const float* bnv,
                float* out,int Hh,int Ww,int Cin,int k,int Cout,int Ctot,int c0,int act){
    int taps=k*k, CoutT=(Cout+15)/16, CoutP=CoutT*16;
    long long wtot=(long long)CoutP*taps*CinP;
    k_pack_w<<<blk(wtot),256,0,stream>>>(wsrc,wslab,Cout,Cin,taps,CoutP,CinP);
    k_bnprep<<<1,64,0,stream>>>(bng,bnb,bnm,bnv,bias,scsl,shsl,Cout, bng?1:0);
    dim3 grid(Ww/32, Hh, BB*CoutT);
    k_conv_wmma<<<grid,64,0,stream>>>(feat,wslab,scsl,shsl,out,BB,Hh,Ww,CinP,taps,
                                      Cout,Ctot,c0,act);
  };
  // Guided Upsampling Block. pi = base leaf index of its params.
  auto gub=[&](int pi,const float* guide,int Cg,const float* depth,int Cd,
               int expn,int outf,float* outbuf,int outCtot,int outC0,int Hh,int Ww){
    long long HW=(long long)Hh*Ww; int comb=expn;
    packf(depth,fsl,BB,HW,Cd,32);                                              // fc1
    conv(fsl,32,F(pi+0),F(pi+1),F(pi+2),F(pi+3),F(pi+4),F(pi+5), sA,Hh,Ww,Cd,3,expn,expn,0,1);
    packf(sA,fsl,BB,HW,expn,32);                                               // fc2
    conv(fsl,32,F(pi+6),F(pi+7),F(pi+8),F(pi+9),F(pi+10),F(pi+11), sXY,Hh,Ww,expn,1,expn/2,comb,0,1);
    packf(guide,fsl,BB,HW,Cg,32);                                              // gc1
    conv(fsl,32,F(pi+12),F(pi+13),F(pi+14),F(pi+15),F(pi+16),F(pi+17), sA,Hh,Ww,Cg,3,expn,expn,0,1);
    packf(sA,fsl,BB,HW,expn,32);                                               // gc2
    conv(fsl,32,F(pi+18),F(pi+19),F(pi+20),F(pi+21),F(pi+22),F(pi+23), sXY,Hh,Ww,expn,1,expn/2,comb,comb/2,1);
    k_gap<<<dim3(comb,BB),256,0,stream>>>(sXY,sev,(int)HW);                    // SE
    k_se<<<BB,32,0,stream>>>(sev,F(pi+38),F(pi+39),seo,comb);
    long long tot=(long long)BB*comb*HW;
    k_chan_scale<<<blk(tot),256,0,stream>>>(sXY,seo,(int)HW,tot);
    packf(sXY,fsl,BB,HW,comb,32);                                              // cc1
    conv(fsl,32,F(pi+24),F(pi+25),F(pi+26),F(pi+27),F(pi+28),F(pi+29), sA,Hh,Ww,comb,3,expn,expn,0,1);
    packf(sA,fsl,BB,HW,expn,32);                                               // cc2
    conv(fsl,32,F(pi+30),F(pi+31),F(pi+32),F(pi+33),F(pi+34),F(pi+35), sB,Hh,Ww,expn,1,Cd,Cd,0,1);
    long long atot=(long long)BB*Cd*HW;                                        // r + depth
    k_add<<<blk(atot),256,0,stream>>>(sB,depth,atot);
    packf(sB,fsl,BB,HW,Cd,32);                                                 // red (bias only)
    conv(fsl,32,F(pi+36),F(pi+37),nullptr,nullptr,nullptr,nullptr, outbuf,Hh,Ww,Cd,1,outf,outCtot,outC0,0);
  };
  auto convbn=[&](const f16* feat,int Cin,int pb,int Hh,int Ww,int Cout,float* out){
    conv(feat,Cin,F(pb+0),nullptr,F(pb+1),F(pb+2),F(pb+3),F(pb+4),out,Hh,Ww,Cin,3,Cout,Cout,0,0);
  };

  // leaf index map (setup_inputs insertion order, params flattened recursively)
  // NOTE: IX_ prefix to avoid HIP macro collisions (MASK2 etc. are macros in HIP headers)
  enum { IX_GUB_RGB=7, IX_GUB_RGB_H=47, IX_GUB_SP=87, IX_GUB_SP_H=127,
         IX_MASK=167, IX_KC=172, IX_G3=177, IX_G5=182, IX_G7=187,
         IX_MASK2=192, IX_KC2=197, IX_G3_2=202, IX_G5_2=207, IX_G7_2=212 };

  // ---- GUB phase: build feat2 (128ch half) and feat1 (64ch full) via concat c0 ----
  gub(IX_GUB_RGB_H, F(0),3, F(2),32, 32,64, feat2,128,0,  H2,W2);  // guide=rgb_half, depth=y_half
  gub(IX_GUB_SP_H,  F(4),1, F(2),32, 32,64, feat2,128,64, H2,W2);  // guide=sparse_half
  gub(IX_GUB_RGB,   F(1),3, F(3),16, 16,32, feat1,64,0,   HH,WW);  // guide=rgb, depth=y
  gub(IX_GUB_SP,    F(5),1, F(3),16, 16,32, feat1,64,32,  HH,WW);  // guide=sparse

  packf(feat1,feat1h,BB,HW1,64,64);
  packf(feat2,feat2h,BB,HW2,128,128);

  // ---- validity + sparse downsample ----
  k_valid<<<blk(P1),256,0,stream>>>(F(5),validb,P1);
  k_sparse_down<<<blk(P2),256,0,stream>>>(F(5),validb,ds2,vm2,HH,WW,BB);

  // ---- half-res heads (feat2, 128ch) ----
  convbn(feat2h,128,IX_MASK2,H2,W2,1, mask2b);
  k_sigmoid_mul<<<blk(P2),256,0,stream>>>(mask2b,vm2,mask2b,P2);
  convbn(feat2h,128,IX_KC2,H2,W2,3, kconf2);
  k_softmax3<<<blk(P2),256,0,stream>>>(kconf2,(int)HW2,BB);
  convbn(feat2h,128,IX_G3_2,H2,W2,8, rawS);
  k_cspn_norm<<<blk(P2),256,0,stream>>>(rawS,normS,9,(int)HW2,BB);
  k_ktrans<<<blk((long long)BB*9*HW2),256,0,stream>>>(normS,g3h,3,H2,W2,BB);
  convbn(feat2h,128,IX_G5_2,H2,W2,24, rawS);
  k_cspn_norm<<<blk(P2),256,0,stream>>>(rawS,normS,25,(int)HW2,BB);
  k_ktrans<<<blk((long long)BB*25*HW2),256,0,stream>>>(normS,g5h,5,H2,W2,BB);
  convbn(feat2h,128,IX_G7_2,H2,W2,48, rawS);
  k_cspn_norm<<<blk(P2),256,0,stream>>>(rawS,normS,49,(int)HW2,BB);
  k_ktrans<<<blk((long long)BB*49*HW2),256,0,stream>>>(normS,g7h,7,H2,W2,BB);

  // ---- full-res heads (feat1, 64ch) ----
  convbn(feat1h,64,IX_MASK,HH,WW,1, maskb);
  k_sigmoid_mul<<<blk(P1),256,0,stream>>>(maskb,validb,maskb,P1);
  convbn(feat1h,64,IX_KC,HH,WW,3, kconf);
  k_softmax3<<<blk(P1),256,0,stream>>>(kconf,(int)HW1,BB);
  convbn(feat1h,64,IX_G3,HH,WW,8, rawS);
  k_cspn_norm<<<blk(P1),256,0,stream>>>(rawS,normS,9,(int)HW1,BB);
  k_ktrans<<<blk((long long)BB*9*HW1),256,0,stream>>>(normS,g3b,3,HH,WW,BB);
  convbn(feat1h,64,IX_G5,HH,WW,24, rawS);
  k_cspn_norm<<<blk(P1),256,0,stream>>>(rawS,normS,25,(int)HW1,BB);
  k_ktrans<<<blk((long long)BB*25*HW1),256,0,stream>>>(normS,g5b,5,HH,WW,BB);
  convbn(feat1h,64,IX_G7,HH,WW,48, rawS);
  k_cspn_norm<<<blk(P1),256,0,stream>>>(rawS,normS,49,(int)HW1,BB);
  k_ktrans<<<blk((long long)BB*49*HW1),256,0,stream>>>(normS,g7b,7,HH,WW,BB);

  // ---- stage-2 (dilated) CSPN step + blend; half-res kernels sampled in place ----
  const float* pred=F(6);
  k_cspn_blend<<<blk(P1),256,0,stream>>>(g3h,pred,pred,mask2b,ds2,t3,3,2,2,HH,WW,BB,1);
  k_cspn_blend<<<blk(P1),256,0,stream>>>(g5h,pred,pred,mask2b,ds2,t5,5,2,4,HH,WW,BB,1);
  k_cspn_blend<<<blk(P1),256,0,stream>>>(g7h,pred,pred,mask2b,ds2,t7,7,2,6,HH,WW,BB,1);
  k_blend3<<<blk(P1),256,0,stream>>>(kconf2,t3,t5,t7,dep2,HH,WW,BB,1);

  // ---- stage-1 CSPN step + blend ----
  k_cspn_blend<<<blk(P1),256,0,stream>>>(g3b,dep2,dep2,maskb,F(5),t3,3,1,1,HH,WW,BB,0);
  k_cspn_blend<<<blk(P1),256,0,stream>>>(g5b,dep2,dep2,maskb,F(5),t5,5,1,2,HH,WW,BB,0);
  k_cspn_blend<<<blk(P1),256,0,stream>>>(g7b,dep2,dep2,maskb,F(5),t7,7,1,3,HH,WW,BB,0);
  k_blend3<<<blk(P1),256,0,stream>>>(kconf,t3,t5,t7,(float*)d_out,HH,WW,BB,0);
}
